// WorkflowEncoder_60979945668776
// MI455X (gfx1250) — matrix operand order repaired and checked by reading, not compiled, not run on hardware
//
#include <hip/hip_runtime.h>

// ---------------------------------------------------------------------------
// CDNA5 (gfx1250, wave32) WMMA implementation of the 3-layer MPNN.
//   - activations/weights in f16, accumulate f32 via v_wmma_f32_16x16x32_f16
//   - weights pre-transposed to [N,K] f16 so B-fragments are contiguous loads
//   - fused edge kernel: gather cat([x[src],x[dst]]) -> LDS -> GEMM1(relu)
//     -> LDS -> GEMM2 -> atomic scatter-add into agg[dst]
// ---------------------------------------------------------------------------

typedef __attribute__((ext_vector_type(16))) _Float16 v16h;
typedef __attribute__((ext_vector_type(8)))  _Float16 v8h;
typedef __attribute__((ext_vector_type(8)))  float    v8f;

#define HDIM 256

// Wave-level K-loop: one wave accumulates a 16x128 output stripe (8 16x16 tiles).
// A (16 rows x KD) comes from LDS, W^T ([256][KD], f16) from global.
// A-layout used here (matches B): lane l -> row l%16; elems 0..15 -> K = kb + (l<16?0:16)+e.
// B-layout used here (matches A): lane l -> col l%16; elems 0..15 -> K = kb + (l<16?0:16)+e.
template <int KD>
__device__ inline void wave_gemm_acc(const _Float16* sA, const _Float16* __restrict__ WT,
                                     int rblk, int colBase, int lrow, int hi, v8f acc[8]) {
    for (int kb = 0; kb < KD; kb += 32) {
        union { v16h v; v8h h[2]; } af;
        const _Float16* ap = sA + (rblk * 16 + lrow) * KD + kb + hi * 16;
        af.h[0] = *(const v8h*)(ap);
        af.h[1] = *(const v8h*)(ap + 8);
#pragma unroll
        for (int t = 0; t < 8; t++) {
            const v16h b = *(const v16h*)(WT + (colBase + t * 16 + lrow) * KD + kb + hi * 16);
            acc[t] = __builtin_amdgcn_wmma_f32_16x16x32_f16(
                false, af.v, false, b, (short)0, acc[t], false, false);
        }
    }
}

// C[M,256] = act(A_f16[M,KD] @ W^T_f16[256,KD] + bias); optional f32 and f16 outputs.
template <int KD>
__global__ __launch_bounds__(256) void k_gemm(const _Float16* __restrict__ A, int M,
                                              const _Float16* __restrict__ WT,
                                              const float* __restrict__ bias,
                                              float* __restrict__ Cf,
                                              _Float16* __restrict__ Ch, int relu) {
    __shared__ _Float16 sA[64 * KD];
    const int tid = threadIdx.x;
    const int rowBase = blockIdx.x * 64;
    constexpr int CPR = KD / 8;  // 16B chunks per row
    for (int c = tid; c < 64 * CPR; c += 256) {
        int r = c / CPR, o = c % CPR;
        int gr = rowBase + r;
        v8h v = {};
        if (gr < M) v = *(const v8h*)(A + (size_t)gr * KD + o * 8);
        *(v8h*)(&sA[r * KD + o * 8]) = v;
    }
    __syncthreads();

    const int wid = tid >> 5, lane = tid & 31;
    const int rblk = wid & 3, colBase = (wid >> 2) * 128;
    const int lrow = lane & 15, hi = lane >> 4;

    const v8f z = {};
    v8f acc[8];
#pragma unroll
    for (int t = 0; t < 8; t++) acc[t] = z;

    wave_gemm_acc<KD>(sA, WT, rblk, colBase, lrow, hi, acc);

    const int r0 = rowBase + rblk * 16 + hi * 8;  // C/D: M = vgpr + (lane<16?0:8)
#pragma unroll
    for (int t = 0; t < 8; t++) {
        int col = colBase + t * 16 + lrow;
        float bv = bias[col];
#pragma unroll
        for (int v = 0; v < 8; v++) {
            int row = r0 + v;
            if (row < M) {
                float x = acc[t][v] + bv;
                if (relu) x = fmaxf(x, 0.f);
                if (Cf) Cf[(size_t)row * HDIM + col] = x;
                if (Ch) Ch[(size_t)row * HDIM + col] = (_Float16)x;
            }
        }
    }
}

// Fused per-edge message MLP + scatter-add. 64 edges per workgroup.
__global__ __launch_bounds__(256) void k_edge_msg(const _Float16* __restrict__ xh,
                                                  const int* __restrict__ src,
                                                  const int* __restrict__ dst,
                                                  const _Float16* __restrict__ w1T,  // [256][512]
                                                  const float* __restrict__ b1,
                                                  const _Float16* __restrict__ w2T,  // [256][256]
                                                  const float* __restrict__ b2,
                                                  float* __restrict__ agg, int Etot) {
    __shared__ _Float16 sA[64 * 512];  // gathered cat([x[src], x[dst]])
    __shared__ _Float16 sH[64 * 256];  // relu hidden
    __shared__ int sDst[64];

    const int tid = threadIdx.x;
    const int eBase = blockIdx.x * 64;
    if (tid < 64) {
        int e = eBase + tid;
        sDst[tid] = dst[(e < Etot) ? e : (Etot - 1)];
    }
    // Gather: 64 rows x 64 chunks of 16B (first 32 chunks from src, next from dst).
    for (int c = tid; c < 64 * 64; c += 256) {
        int r = c >> 6, o = c & 63;
        int e = eBase + r;
        if (e >= Etot) e = Etot - 1;
        int node = (o < 32) ? src[e] : dst[e];
        *(v8h*)(&sA[r * 512 + o * 8]) = *(const v8h*)(xh + (size_t)node * HDIM + (o & 31) * 8);
    }
    __syncthreads();

    const int wid = tid >> 5, lane = tid & 31;
    const int rblk = wid & 3, colBase = (wid >> 2) * 128;
    const int lrow = lane & 15, hi = lane >> 4;
    const v8f z = {};

    {  // GEMM1: hid = relu(cat @ W1 + b1) -> LDS (f16)
        v8f acc[8];
#pragma unroll
        for (int t = 0; t < 8; t++) acc[t] = z;
        wave_gemm_acc<512>(sA, w1T, rblk, colBase, lrow, hi, acc);
#pragma unroll
        for (int t = 0; t < 8; t++) {
            int col = colBase + t * 16 + lrow;
            float bv = b1[col];
#pragma unroll
            for (int v = 0; v < 8; v++) {
                int row = rblk * 16 + hi * 8 + v;
                sH[row * HDIM + col] = (_Float16)fmaxf(acc[t][v] + bv, 0.f);
            }
        }
    }
    __syncthreads();
    {  // GEMM2: msg = hid @ W2 + b2, then segment-sum via atomics
        v8f acc[8];
#pragma unroll
        for (int t = 0; t < 8; t++) acc[t] = z;
        wave_gemm_acc<256>(sH, w2T, rblk, colBase, lrow, hi, acc);
#pragma unroll
        for (int t = 0; t < 8; t++) {
            int col = colBase + t * 16 + lrow;
            float bv = b2[col];
#pragma unroll
            for (int v = 0; v < 8; v++) {
                int row = rblk * 16 + hi * 8 + v;
                if (eBase + row < Etot)
                    atomicAdd(&agg[(size_t)sDst[row] * HDIM + col], acc[t][v] + bv);
            }
        }
    }
}

// W [K][256] f32 (row-major) -> W^T [256][K] f16
__global__ void k_wt(const float* __restrict__ in, _Float16* __restrict__ out, int K) {
    int i = blockIdx.x * 256 + threadIdx.x;
    if (i < K * 256) {
        int k = i >> 8, n = i & 255;
        out[n * K + k] = (_Float16)in[i];
    }
}

__global__ void k_cvt(const float* __restrict__ in, _Float16* __restrict__ out, int n) {
    int i = blockIdx.x * 256 + threadIdx.x;
    if (i < n) out[i] = (_Float16)in[i];
}

__global__ void k_zero(float* p, int n) {
    int i = blockIdx.x * 256 + threadIdx.x;
    if (i < n) p[i] = 0.f;
}

// cat_n = [x_f16 | (f16)agg]
__global__ void k_concat(const _Float16* __restrict__ xh, const float* __restrict__ agg,
                         _Float16* __restrict__ cat, int total) {
    int i = blockIdx.x * 256 + threadIdx.x;
    if (i < total) {
        int n = i >> 8, c = i & 255;
        cat[(size_t)n * 512 + c] = xh[i];
        cat[(size_t)n * 512 + 256 + c] = (_Float16)agg[i];
    }
}

// x = LayerNorm(x + upd) * g + b  (one block per node), writes f32 + f16 copies
__global__ __launch_bounds__(256) void k_ln(const float* __restrict__ x,
                                            const float* __restrict__ upd,
                                            const float* __restrict__ g,
                                            const float* __restrict__ b,
                                            float* __restrict__ xo,
                                            _Float16* __restrict__ xho) {
    __shared__ float s[256];
    const int n = blockIdx.x, c = threadIdx.x;
    float r = x[(size_t)n * HDIM + c] + upd[(size_t)n * HDIM + c];
    s[c] = r;
    __syncthreads();
    for (int st = 128; st > 0; st >>= 1) { if (c < st) s[c] += s[c + st]; __syncthreads(); }
    float mu = s[0] * (1.f / HDIM);
    __syncthreads();
    float d = r - mu;
    s[c] = d * d;
    __syncthreads();
    for (int st = 128; st > 0; st >>= 1) { if (c < st) s[c] += s[c + st]; __syncthreads(); }
    float var = s[0] * (1.f / HDIM);
    float y = d * rsqrtf(var + 1e-5f) * g[c] + b[c];
    xo[(size_t)n * HDIM + c] = y;
    xho[(size_t)n * HDIM + c] = (_Float16)y;
}

__global__ __launch_bounds__(256) void k_colsum(const float* __restrict__ x,
                                                float* __restrict__ gsum, int Nn) {
    float a = 0.f;
    for (int n = blockIdx.x; n < Nn; n += gridDim.x) a += x[(size_t)n * HDIM + threadIdx.x];
    atomicAdd(&gsum[threadIdx.x], a);
}

// Tiny readout MLP on the mean vector (single block, f32 weights direct).
__global__ __launch_bounds__(256) void k_readout(const float* __restrict__ gsum,
                                                 const float* __restrict__ w1,
                                                 const float* __restrict__ b1,
                                                 const float* __restrict__ w2,
                                                 const float* __restrict__ b2,
                                                 float* __restrict__ out, float invN) {
    __shared__ float gm[256], h[256];
    const int c = threadIdx.x;
    gm[c] = gsum[c] * invN;
    __syncthreads();
    float a = b1[c];
    for (int k = 0; k < 256; k++) a += gm[k] * w1[k * 256 + c];
    h[c] = fmaxf(a, 0.f);
    __syncthreads();
    float o = b2[c];
    for (int k = 0; k < 256; k++) o += h[k] * w2[k * 256 + c];
    out[c] = o;
}

extern "C" void kernel_launch(void* const* d_in, const int* in_sizes, int n_in,
                              void* d_out, int out_size, void* d_ws, size_t ws_size,
                              hipStream_t stream) {
    (void)n_in; (void)out_size; (void)ws_size;
    constexpr int H = 256, L = 3, TWO_H = 512;
    const float* nf     = (const float*)d_in[0];
    const int*   ei     = (const int*)d_in[1];  // [2,E] (JAX default int32)
    const float* ne_w1  = (const float*)d_in[2];
    const float* ne_b1  = (const float*)d_in[3];
    const float* ne_w2  = (const float*)d_in[4];
    const float* ne_b2  = (const float*)d_in[5];
    const float* msg_w1 = (const float*)d_in[6];
    const float* msg_b1 = (const float*)d_in[7];
    const float* msg_w2 = (const float*)d_in[8];
    const float* msg_b2 = (const float*)d_in[9];
    const float* upd_w1 = (const float*)d_in[10];
    const float* upd_b1 = (const float*)d_in[11];
    const float* upd_w2 = (const float*)d_in[12];
    const float* upd_b2 = (const float*)d_in[13];
    const float* ln_g   = (const float*)d_in[14];
    const float* ln_b   = (const float*)d_in[15];
    const float* ro_w1  = (const float*)d_in[16];
    const float* ro_b1  = (const float*)d_in[17];
    const float* ro_w2  = (const float*)d_in[18];
    const float* ro_b2  = (const float*)d_in[19];
    float* out = (float*)d_out;

    const int Nn = in_sizes[0] / H;
    const int E  = in_sizes[1] / 2;
    const int* srcI = ei;
    const int* dstI = ei + E;

    char* ws = (char*)d_ws;
    size_t off = 0;
    auto alloc = [&](size_t b) -> void* {
        void* p = ws + off;
        off += (b + 255) & ~(size_t)255;
        return p;
    };
    float*    xf32  = (float*)alloc((size_t)Nn * H * 4);      // node state f32
    float*    agg   = (float*)alloc((size_t)Nn * H * 4);      // aggregate / upd f32
    _Float16* xh    = (_Float16*)alloc((size_t)Nn * H * 2);   // node state f16
    _Float16* hid   = (_Float16*)alloc((size_t)Nn * H * 2);   // MLP hidden f16
    _Float16* cat16 = (_Float16*)alloc((size_t)Nn * TWO_H * 2);
    float*    gsum  = (float*)alloc(H * 4);
    _Float16* neW1T = (_Float16*)alloc((size_t)H * H * 2);
    _Float16* neW2T = (_Float16*)alloc((size_t)H * H * 2);
    _Float16* mW1T  = (_Float16*)alloc((size_t)L * TWO_H * H * 2);
    _Float16* mW2T  = (_Float16*)alloc((size_t)L * H * H * 2);
    _Float16* uW1T  = (_Float16*)alloc((size_t)L * TWO_H * H * 2);
    _Float16* uW2T  = (_Float16*)alloc((size_t)L * H * H * 2);

    auto cdiv = [](int a, int b) { return (a + b - 1) / b; };
    const dim3 B256(256);

    // ---- weight transpose + f16 convert (L2-resident afterwards) ----
    k_wt<<<cdiv(H * H, 256), B256, 0, stream>>>(ne_w1, neW1T, H);
    k_wt<<<cdiv(H * H, 256), B256, 0, stream>>>(ne_w2, neW2T, H);
    for (int l = 0; l < L; l++) {
        k_wt<<<cdiv(TWO_H * H, 256), B256, 0, stream>>>(msg_w1 + (size_t)l * TWO_H * H,
                                                        mW1T + (size_t)l * H * TWO_H, TWO_H);
        k_wt<<<cdiv(H * H, 256), B256, 0, stream>>>(msg_w2 + (size_t)l * H * H,
                                                    mW2T + (size_t)l * H * H, H);
        k_wt<<<cdiv(TWO_H * H, 256), B256, 0, stream>>>(upd_w1 + (size_t)l * TWO_H * H,
                                                        uW1T + (size_t)l * H * TWO_H, TWO_H);
        k_wt<<<cdiv(H * H, 256), B256, 0, stream>>>(upd_w2 + (size_t)l * H * H,
                                                    uW2T + (size_t)l * H * H, H);
    }

    // ---- node encoder: x = relu(nf@W1+b1)@W2+b2 ----
    k_cvt<<<cdiv(Nn * H, 256), B256, 0, stream>>>(nf, xh, Nn * H);
    k_gemm<256><<<cdiv(Nn, 64), B256, 0, stream>>>(xh, Nn, neW1T, ne_b1, nullptr, hid, 1);
    k_gemm<256><<<cdiv(Nn, 64), B256, 0, stream>>>(hid, Nn, neW2T, ne_b2, xf32, xh, 0);

    // ---- message passing layers ----
    for (int l = 0; l < L; l++) {
        k_zero<<<cdiv(Nn * H, 256), B256, 0, stream>>>(agg, Nn * H);
        k_edge_msg<<<cdiv(E, 64), B256, 0, stream>>>(xh, srcI, dstI,
                                                     mW1T + (size_t)l * H * TWO_H, msg_b1 + l * H,
                                                     mW2T + (size_t)l * H * H, msg_b2 + l * H,
                                                     agg, E);
        k_concat<<<cdiv(Nn * H, 256), B256, 0, stream>>>(xh, agg, cat16, Nn * H);
        k_gemm<512><<<cdiv(Nn, 64), B256, 0, stream>>>(cat16, Nn, uW1T + (size_t)l * H * TWO_H,
                                                       upd_b1 + l * H, nullptr, hid, 1);
        k_gemm<256><<<cdiv(Nn, 64), B256, 0, stream>>>(hid, Nn, uW2T + (size_t)l * H * H,
                                                       upd_b2 + l * H, agg, nullptr, 0);
        k_ln<<<Nn, B256, 0, stream>>>(xf32, agg, ln_g + l * H, ln_b + l * H, xf32, xh);
    }

    // ---- mean readout + final MLP ----
    k_zero<<<1, B256, 0, stream>>>(gsum, H);
    k_colsum<<<256, B256, 0, stream>>>(xf32, gsum, Nn);
    k_readout<<<1, B256, 0, stream>>>(gsum, ro_w1, ro_b1, ro_w2, ro_b2, out, 1.0f / (float)Nn);
}
